// Scaled_DotProduct_Attention_19739669692946
// MI455X (gfx1250) — matrix operand order repaired and checked by reading, not compiled, not run on hardware
//
#include <hip/hip_runtime.h>
#include <hip/hip_bf16.h>
#include <stddef.h>

// Attention variant: softmax over the HEADS axis (reference: softmax(dim=1)).
//   B=2, H=16, S=2048, D=64.
//   W[b,h,q,k] = exp(s)/sum_h' exp(s'),  s = (Q.K)/sqrt(D), masked -> 0
//   out[b,q,h,d] = sum_k W[b,h,q,k] * V[b,h,k,d]
//
// Pass 0: Qh = f16(Q * (1/8) * log2(e)), Kh = f16(K), Vt = f16(V^T) [bh][d][k]
//         (scores come out of WMMA already in log2 domain -> v_exp_f32 direct)
// Pass 1: drcp[b,q,k] = 1 / sum_h exp2(masked score)          (33.5 MB ws)
// Pass 2: per (b,h,q-tile) wave: recompute scores with WMMA, normalize,
//         ctx += W @ V with WMMA (V^T fragments are contiguous loads).
// All matmuls: v_wmma_f32_16x16x32_f16 (wave32 CDNA5).

typedef __attribute__((ext_vector_type(16))) _Float16 v16h;
typedef __attribute__((ext_vector_type(8)))  _Float16 v8h;
typedef __attribute__((ext_vector_type(8)))  float    v8f;
typedef __attribute__((ext_vector_type(4)))  float    v4f;

#define ATT_B 2
#define ATT_H 16
#define ATT_S 2048
#define ATT_D 64

__device__ __forceinline__ float fast_exp2(float x) {
  return __builtin_amdgcn_exp2f(x);  // v_exp_f32 (natively 2^x)
}

__device__ __forceinline__ v8f wmma_f16(v16h a, v16h b, v8f c) {
  return __builtin_amdgcn_wmma_f32_16x16x32_f16(
      false, a, false, b, (short)0, c, false, false);
}

// A-fragment (16 rows x 32 k) from row-major f16 tile, ld = 64.
// lane half=0 holds K = d0+0..15 ; half=1: d0+16..31 (contiguous order).
__device__ __forceinline__ v16h load_a_h(const _Float16* __restrict__ tile,
                                         int lane, int d0) {
  const int r = lane & 15, half = lane >> 4;
  const _Float16* p = tile + r * ATT_D + d0 + half * 16;
  const v8h lo = *(const v8h*)p;
  const v8h hi = *(const v8h*)(p + 8);
  v16h a{};
#pragma unroll
  for (int i = 0; i < 8; ++i) { a[i] = lo[i]; a[i + 8] = hi[i]; }
  return a;
}

// byte != 0 -> 0.0f gate, branch-free (v_cmp + v_cndmask)
__device__ __forceinline__ float mgate(unsigned int word, int byte) {
  return ((word >> (8 * byte)) & 0xffu) ? 0.f : 1.f;
}

// ---------------- Pass 0a: scale + f32->f16 convert (row-major copy) ------
__global__ __launch_bounds__(256) void cvt_kernel(
    const float* __restrict__ src, _Float16* __restrict__ dst, float scale) {
  const size_t i = ((size_t)blockIdx.x * 256 + threadIdx.x) * 8;
  const v4f a = *(const v4f*)(src + i);
  const v4f b = *(const v4f*)(src + i + 4);
  v8h o{};
#pragma unroll
  for (int j = 0; j < 4; ++j) {
    o[j]     = (_Float16)(a[j] * scale);
    o[j + 4] = (_Float16)(b[j] * scale);
  }
  *(v8h*)(dst + i) = o;
}

// ---------------- Pass 0b: V -> f16 V^T  (Vt[bh][d][k]) -------------------
__global__ __launch_bounds__(256) void vtrans_kernel(
    const float* __restrict__ V, _Float16* __restrict__ Vt) {
  const int idx = blockIdx.x * 256 + threadIdx.x;  // 32*64*256 = 524288
  const int k8 = (idx & 255) * 8;
  const int d  = (idx >> 8) & 63;
  const int bh = idx >> 14;
  const float* src = V + ((size_t)bh * ATT_S + k8) * ATT_D + d;
  _Float16* dst = Vt + ((size_t)bh * ATT_D + d) * ATT_S + k8;
  v8h o{};
#pragma unroll
  for (int i = 0; i < 8; ++i) o[i] = (_Float16)src[(size_t)i * ATT_D];
  *(v8h*)dst = o;
}

// ---------------- Pass 1: drcp[b,q,k] = 1 / sum_h exp2 --------------------
__global__ __launch_bounds__(512) void attn_denom_kernel(
    const _Float16* __restrict__ Qh, const _Float16* __restrict__ Kh,
    const unsigned char* __restrict__ M, float* __restrict__ drcp) {
  __shared__ float red[ATT_H * 256];

  const int b = blockIdx.z, qt = blockIdx.y, kt = blockIdx.x;
  const int h = threadIdx.x >> 5, lane = threadIdx.x & 31;
  const int n = lane & 15, half = lane >> 4;
  const int bh = b * ATT_H + h;

  const _Float16* Qt = Qh + ((size_t)bh * ATT_S + (size_t)qt * 16) * ATT_D;
  const _Float16* Kt = Kh + ((size_t)bh * ATT_S + (size_t)kt * 16) * ATT_D;

  // B = K^T: lane n holds K-tile row n; contraction d contiguous (32B loads)
  v8f c = {};
  c = wmma_f16(load_a_h(Qt, lane, 0),
               *(const v16h*)(Kt + (size_t)n * ATT_D + half * 16), c);
  c = wmma_f16(load_a_h(Qt, lane, 32),
               *(const v16h*)(Kt + (size_t)n * ATT_D + 32 + half * 16), c);

  // scores -> LDS in C-layout
#pragma unroll
  for (int j = 0; j < 8; ++j) red[h * 256 + (j + 8 * half) * 16 + n] = c[j];
  __syncthreads();

  // masked exp2 in row layout: lane owns 8 contiguous k of one row
  {
    const int r = lane >> 1, c0 = (lane & 1) * 8;
    float* slot = &red[h * 256 + r * 16 + c0];
    const unsigned char* mrow =
        M + (((size_t)bh * ATT_S + (size_t)qt * 16 + r) * ATT_S) +
        (size_t)kt * 16 + c0;
    const uint2 m8 = *(const uint2*)mrow;
    const v4f lo = *(const v4f*)slot;
    const v4f hi = *(const v4f*)(slot + 4);
    v4f wlo{}, whi{};
#pragma unroll
    for (int i = 0; i < 4; ++i) {
      wlo[i] = mgate(m8.x, i) * fast_exp2(lo[i]);
      whi[i] = mgate(m8.y, i) * fast_exp2(hi[i]);
    }
    *(v4f*)slot = wlo;
    *(v4f*)(slot + 4) = whi;
  }
  __syncthreads();

  const int t = threadIdx.x;
  if (t < 256) {
    float s = 0.f;
#pragma unroll
    for (int hh = 0; hh < ATT_H; ++hh) s += red[hh * 256 + t];
    drcp[((size_t)b * ATT_S + (size_t)qt * 16 + (t >> 4)) * ATT_S +
         (size_t)kt * 16 + (t & 15)] = 1.0f / s;
  }
}

// ---------------- Pass 2: context = W @ V, out[b,q,h,d] -------------------
__global__ __launch_bounds__(256) void attn_ctx_kernel(
    const _Float16* __restrict__ Qh, const _Float16* __restrict__ Kh,
    const _Float16* __restrict__ Vt, const unsigned char* __restrict__ M,
    const float* __restrict__ drcp, float* __restrict__ out) {
  __shared__ float Sbuf[8][16 * 32];  // per-wave score tile (16 q x 32 k)

  const int wid = threadIdx.x >> 5, lane = threadIdx.x & 31;
  const int n = lane & 15, half = lane >> 4;

  const int task = blockIdx.x * 8 + wid;  // 0 .. 4095
  const int qt = task & 127;              // S/16
  const int bh = task >> 7;
  const int h = bh & 15, b = bh >> 4;

  const _Float16* Qt = Qh + ((size_t)bh * ATT_S + (size_t)qt * 16) * ATT_D;
  const _Float16* Kb = Kh + (size_t)bh * ATT_S * ATT_D;
  const _Float16* Vb = Vt + (size_t)bh * ATT_D * ATT_S;
  const unsigned char* Mb =
      M + ((size_t)bh * ATT_S + (size_t)qt * 16) * ATT_S;
  const float* Db = drcp + ((size_t)b * ATT_S + (size_t)qt * 16) * ATT_S;

  const v16h aq0 = load_a_h(Qt, lane, 0);
  const v16h aq1 = load_a_h(Qt, lane, 32);

  v8f ctx0 = {}, ctx1 = {}, ctx2 = {}, ctx3 = {};
  float* swave = &Sbuf[wid][0];
  const int row = n;

  for (int kc = 0; kc < ATT_S; kc += 32) {
    // --- scores: B fragments = contiguous 32B rows of Kh ---
    const _Float16* kcol0 = Kb + (size_t)(kc + n) * ATT_D + half * 16;
    const _Float16* kcol1 = Kb + (size_t)(kc + 16 + n) * ATT_D + half * 16;
    v8f s0 = {}, s1 = {};
    s0 = wmma_f16(aq0, *(const v16h*)kcol0, s0);
    s0 = wmma_f16(aq1, *(const v16h*)(kcol0 + 32), s0);
    s1 = wmma_f16(aq0, *(const v16h*)kcol1, s1);
    s1 = wmma_f16(aq1, *(const v16h*)(kcol1 + 32), s1);

    if (kc + 32 < ATT_S) {  // prefetch next chunk (global_prefetch_b8)
      __builtin_prefetch(kcol0 + 2 * ATT_D * 16, 0, 1);
      __builtin_prefetch(Vb + (size_t)n * ATT_S + kc + 32, 0, 1);
    }

    // --- C-layout -> LDS ---
#pragma unroll
    for (int j = 0; j < 8; ++j) {
      const int m = j + 8 * half;
      swave[m * 32 + n] = s0[j];
      swave[m * 32 + 16 + n] = s1[j];
    }
    __syncthreads();

    // --- build f16 A-fragment of W in row layout (contiguous mask/denom) --
    v16h aw{};
    {
      const float* srow = swave + row * 32;
      const v4f sv0 = *(const v4f*)(srow + half * 16);
      const v4f sv1 = *(const v4f*)(srow + half * 16 + 4);
      const v4f sv2 = *(const v4f*)(srow + 8 + half * 16);
      const v4f sv3 = *(const v4f*)(srow + 12 + half * 16);
      const unsigned char* mrow = Mb + (size_t)row * ATT_S + kc;
      const uint2 mlo = *(const uint2*)(mrow + half * 16);
      const uint2 mhi = *(const uint2*)(mrow + 8 + half * 16);
      const float* drow = Db + (size_t)row * ATT_S + kc;
      const v4f dv0 = *(const v4f*)(drow + half * 16);
      const v4f dv1 = *(const v4f*)(drow + half * 16 + 4);
      const v4f dv2 = *(const v4f*)(drow + 8 + half * 16);
      const v4f dv3 = *(const v4f*)(drow + 12 + half * 16);
#pragma unroll
      for (int i = 0; i < 4; ++i) {
        aw[i]      = (_Float16)(mgate(mlo.x, i) * fast_exp2(sv0[i]) * dv0[i]);
        aw[i + 4]  = (_Float16)(mgate(mlo.y, i) * fast_exp2(sv1[i]) * dv1[i]);
        aw[i + 8]  = (_Float16)(mgate(mhi.x, i) * fast_exp2(sv2[i]) * dv2[i]);
        aw[i + 12] = (_Float16)(mgate(mhi.y, i) * fast_exp2(sv3[i]) * dv3[i]);
      }
    }
    __syncthreads();

    // --- ctx += W @ V : B fragments = contiguous 32B rows of V^T ---
    ctx0 = wmma_f16(aw, *(const v16h*)(Vb + (size_t)(0 * 16 + n) * ATT_S +
                                       kc + half * 16), ctx0);
    ctx1 = wmma_f16(aw, *(const v16h*)(Vb + (size_t)(1 * 16 + n) * ATT_S +
                                       kc + half * 16), ctx1);
    ctx2 = wmma_f16(aw, *(const v16h*)(Vb + (size_t)(2 * 16 + n) * ATT_S +
                                       kc + half * 16), ctx2);
    ctx3 = wmma_f16(aw, *(const v16h*)(Vb + (size_t)(3 * 16 + n) * ATT_S +
                                       kc + half * 16), ctx3);
  }

  // --- store out[b, q, h, d] (B,S,H,D) ---
#pragma unroll
  for (int j = 0; j < 8; ++j) {
    const int m = j + 8 * half;
    const size_t q = (size_t)qt * 16 + m;
    float* o = out + (((size_t)b * ATT_S + q) * ATT_H + h) * ATT_D;
    o[0 * 16 + n] = ctx0[j];
    o[1 * 16 + n] = ctx1[j];
    o[2 * 16 + n] = ctx2[j];
    o[3 * 16 + n] = ctx3[j];
  }
}

extern "C" void kernel_launch(void* const* d_in, const int* in_sizes, int n_in,
                              void* d_out, int out_size, void* d_ws, size_t ws_size,
                              hipStream_t stream) {
  const float* Q = (const float*)d_in[0];
  const float* K = (const float*)d_in[1];
  const float* V = (const float*)d_in[2];
  const unsigned char* M = (const unsigned char*)d_in[3];  // jax bool, 1B
  float* out = (float*)d_out;

  // Workspace layout (all offsets 256B-aligned):
  //   [0, 33554432)              : drcp  (B*S*S f32)
  //   [33554432, +8388608)       : Qh    (f16, scaled by log2e/8)
  //   [41943040, +8388608)       : Kh    (f16)
  //   [50331648, +8388608)       : Vt    (f16, transposed [bh][d][k])
  char* ws = (char*)d_ws;
  float*    drcp = (float*)ws;
  _Float16* Qhp  = (_Float16*)(ws + 33554432);
  _Float16* Khp  = (_Float16*)(ws + 41943040);
  _Float16* Vtp  = (_Float16*)(ws + 50331648);

  const int cvtBlocks = (ATT_B * ATT_H * ATT_S * ATT_D) / (256 * 8);  // 2048
  // fold 1/sqrt(D) and log2(e) into Q so scores are exp2-ready
  cvt_kernel<<<cvtBlocks, 256, 0, stream>>>(Q, Qhp,
                                            0.125f * 1.44269504088896340736f);
  cvt_kernel<<<cvtBlocks, 256, 0, stream>>>(K, Khp, 1.0f);
  vtrans_kernel<<<cvtBlocks, 256, 0, stream>>>(V, Vtp);

  dim3 g1(ATT_S / 16, ATT_S / 16, ATT_B);  // (kt, qt, b)
  attn_denom_kernel<<<g1, 512, 0, stream>>>(Qhp, Khp, M, drcp);

  const int tasks = ATT_B * ATT_H * (ATT_S / 16);  // 4096 waves
  attn_ctx_kernel<<<tasks / 8, 256, 0, stream>>>(Qhp, Khp, Vtp, M, drcp, out);
}
